// MambaModel_89996744721114
// MI455X (gfx1250) — matrix operand order repaired and checked by reading, not compiled, not run on hardware
//
#include <hip/hip_runtime.h>

// Problem constants (match reference)
#define BB 2
#define LL 2048
#define DD 1024
#define VV 32000
#define EE 2048
#define NNs 16
#define RR 64
#define KK 4
#define MM (BB * LL)   // 4096 rows of activations

typedef unsigned short u16;
typedef __attribute__((ext_vector_type(8)))  u16   u16x8;
typedef __attribute__((ext_vector_type(16))) u16   u16x16;
typedef __attribute__((ext_vector_type(16))) __bf16 v16bf;
typedef __attribute__((ext_vector_type(8)))  float  v8f;

__device__ __forceinline__ u16 f2bf(float f) {
  unsigned u = __float_as_uint(f);
  u += 0x7FFFu + ((u >> 16) & 1u);          // round-to-nearest-even
  return (u16)(u >> 16);
}
__device__ __forceinline__ float bf2f(u16 h) {
  return __uint_as_float(((unsigned)h) << 16);
}
__device__ __forceinline__ float silu_f(float x) { return x / (1.0f + __expf(-x)); }

// ---------------------------------------------------------------------------
// Generic bf16 WMMA GEMM:  C(MxN,f32) = A(MxK,bf16,row) * Bt(NxK,bf16,row)^T
// Block = 256 thr = 8 waves; wave tile 32x64 (2x4 WMMA frags); block tile 128x128.
// Requires M%128==0, N%128==0 (buffer-padded), K%32==0.
// ---------------------------------------------------------------------------
__device__ __forceinline__ v16bf load_frag_a(const u16* p) {
  u16x8 lo = *(const u16x8*)p;          // K+0..7   (or K+16..23 for hi lanes)
  u16x8 hi = *(const u16x8*)(p + 8);    // K+8..15  (or K+24..31)
  u16x16 u = __builtin_shufflevector(lo, hi, 0,1,2,3,4,5,6,7,8,9,10,11,12,13,14,15);
  return __builtin_bit_cast(v16bf, u);
}
__device__ __forceinline__ v16bf load_frag_b(const u16* p) {
  u16x16 u = *(const u16x16*)p;         // K+0..15 (lanes 0-15) / K+16..31 (lanes 16-31)
  return __builtin_bit_cast(v16bf, u);
}
__device__ __forceinline__ void store_frag_c(float* C, int ldc, int row, int col,
                                             int lane, v8f c, int acc) {
  int cn = lane & 15;
  int mb = (lane >> 4) * 8;
  float* p = C + (size_t)(row + mb) * ldc + (col + cn);
#pragma unroll
  for (int j = 0; j < 8; ++j) {
    float v = c[j];
    if (acc) v += p[(size_t)j * ldc];
    p[(size_t)j * ldc] = v;
  }
}

__global__ void __launch_bounds__(256) gemm_bf16_wmma(
    const u16* __restrict__ A, const u16* __restrict__ Bt, float* __restrict__ C,
    int K, int lda, int ldb, int ldc, int acc)
{
  const int lane = threadIdx.x & 31;
  const int wave = threadIdx.x >> 5;
  const int wy = wave >> 1, wx = wave & 1;
  const int row0 = blockIdx.y * 128 + wy * 32;
  const int col0 = blockIdx.x * 128 + wx * 64;
  const int mrow = lane & 15;
  const int hk   = lane >> 4;   // which K-half this lane holds

  const u16* a0 = A  + (size_t)(row0 + mrow) * lda + hk * 16;
  const u16* a1 = a0 + (size_t)16 * lda;
  const u16* b0 = Bt + (size_t)(col0 + mrow) * ldb + hk * 16;
  const u16* b1 = b0 + (size_t)16 * ldb;
  const u16* b2 = b0 + (size_t)32 * ldb;
  const u16* b3 = b0 + (size_t)48 * ldb;

  v8f c00 = {}, c01 = {}, c02 = {}, c03 = {};
  v8f c10 = {}, c11 = {}, c12 = {}, c13 = {};
  for (int k = 0; k < K; k += 32) {
    v16bf fa0 = load_frag_a(a0 + k);
    v16bf fa1 = load_frag_a(a1 + k);
    v16bf fb0 = load_frag_b(b0 + k);
    v16bf fb1 = load_frag_b(b1 + k);
    v16bf fb2 = load_frag_b(b2 + k);
    v16bf fb3 = load_frag_b(b3 + k);
    c00 = __builtin_amdgcn_wmma_f32_16x16x32_bf16(false, fa0, false, fb0, (short)0, c00, false, false);
    c01 = __builtin_amdgcn_wmma_f32_16x16x32_bf16(false, fa0, false, fb1, (short)0, c01, false, false);
    c02 = __builtin_amdgcn_wmma_f32_16x16x32_bf16(false, fa0, false, fb2, (short)0, c02, false, false);
    c03 = __builtin_amdgcn_wmma_f32_16x16x32_bf16(false, fa0, false, fb3, (short)0, c03, false, false);
    c10 = __builtin_amdgcn_wmma_f32_16x16x32_bf16(false, fa1, false, fb0, (short)0, c10, false, false);
    c11 = __builtin_amdgcn_wmma_f32_16x16x32_bf16(false, fa1, false, fb1, (short)0, c11, false, false);
    c12 = __builtin_amdgcn_wmma_f32_16x16x32_bf16(false, fa1, false, fb2, (short)0, c12, false, false);
    c13 = __builtin_amdgcn_wmma_f32_16x16x32_bf16(false, fa1, false, fb3, (short)0, c13, false, false);
  }
  store_frag_c(C, ldc, row0,      col0,      lane, c00, acc);
  store_frag_c(C, ldc, row0,      col0 + 16, lane, c01, acc);
  store_frag_c(C, ldc, row0,      col0 + 32, lane, c02, acc);
  store_frag_c(C, ldc, row0,      col0 + 48, lane, c03, acc);
  store_frag_c(C, ldc, row0 + 16, col0,      lane, c10, acc);
  store_frag_c(C, ldc, row0 + 16, col0 + 16, lane, c11, acc);
  store_frag_c(C, ldc, row0 + 16, col0 + 32, lane, c12, acc);
  store_frag_c(C, ldc, row0 + 16, col0 + 48, lane, c13, acc);
}

// ---------------------------------------------------------------------------
// Elementwise / support kernels
// ---------------------------------------------------------------------------
__global__ void embed_kernel(const int* __restrict__ tok, const float* __restrict__ emb,
                             float* __restrict__ x) {
  int idx = blockIdx.x * blockDim.x + threadIdx.x;   // m*DD + d
  int d = idx % DD;
  int m = idx / DD;
  x[idx] = emb[(size_t)tok[m] * DD + d];
}

__global__ void rmsnorm_kernel(const float* __restrict__ x, const float* __restrict__ w,
                               u16* __restrict__ o, int Dv) {
  __shared__ float sm[256];
  int row = blockIdx.x;
  const float* xr = x + (size_t)row * Dv;
  float ss = 0.f;
  for (int d = threadIdx.x; d < Dv; d += 256) { float v = xr[d]; ss += v * v; }
  sm[threadIdx.x] = ss;
  __syncthreads();
  for (int s = 128; s > 0; s >>= 1) {
    if ((int)threadIdx.x < s) sm[threadIdx.x] += sm[threadIdx.x + s];
    __syncthreads();
  }
  float scale = rsqrtf(sm[0] / (float)Dv + 1e-5f);
  for (int d = threadIdx.x; d < Dv; d += 256)
    o[(size_t)row * Dv + d] = f2bf(xr[d] * w[d] * scale);
}

// W (Kd x Nd, f32, row-major) -> Bt (Npad x Kd, bf16, row-major), zero-padded rows
__global__ void convT_kernel(u16* __restrict__ out, const float* __restrict__ in,
                             int Kd, int Nd, int Npad) {
  size_t idx = (size_t)blockIdx.x * blockDim.x + threadIdx.x;
  size_t total = (size_t)Npad * Kd;
  if (idx >= total) return;
  int k = (int)(idx % Kd);
  int n = (int)(idx / Kd);
  out[idx] = (n < Nd) ? f2bf(in[(size_t)k * Nd + n]) : (u16)0;
}

__global__ void conv_direct_kernel(u16* __restrict__ out, const float* __restrict__ in,
                                   size_t count) {
  size_t idx = (size_t)blockIdx.x * blockDim.x + threadIdx.x;
  if (idx < count) out[idx] = f2bf(in[idx]);
}

// causal depthwise conv (K=4) + SiLU; reads xc = xz[:, :E] (row stride 2E), writes bf16
__global__ void conv_silu_kernel(const float* __restrict__ xz, const float* __restrict__ cw,
                                 const float* __restrict__ cb, u16* __restrict__ xc) {
  int idx = blockIdx.x * blockDim.x + threadIdx.x;   // m*EE + e
  int e = idx % EE;
  int m = idx / EE;
  int l = m % LL;
  float acc = cb[e];
#pragma unroll
  for (int k = 0; k < KK; ++k) {
    int lp = l + k - (KK - 1);
    if (lp >= 0) acc += cw[e * KK + k] * xz[(size_t)(m - l + lp) * (2 * EE) + e];
  }
  xc[idx] = f2bf(silu_f(acc));
}

// slice dbl[:, 0:R] (ld=128) -> bf16 (M x R) for the dt GEMM
__global__ void slice_kernel(u16* __restrict__ out, const float* __restrict__ dbl) {
  int idx = blockIdx.x * blockDim.x + threadIdx.x;   // m*RR + r
  int r = idx % RR;
  int m = idx / RR;
  out[idx] = f2bf(dbl[(size_t)m * 128 + r]);
}

__global__ void softplus_kernel(float* __restrict__ dt, const float* __restrict__ bias) {
  int idx = blockIdx.x * blockDim.x + threadIdx.x;   // m*EE + e
  int e = idx % EE;
  float v = dt[idx] + bias[e];
  dt[idx] = (v > 20.f) ? v : log1pf(__expf(v));
}

// selective scan: one (b,e) channel per 16 lanes, one state n per lane
__global__ void scan_kernel(const float* __restrict__ dt, const u16* __restrict__ xc,
                            const float* __restrict__ dbl, const float* __restrict__ A_log,
                            float* __restrict__ y) {
  int g = blockIdx.x * (blockDim.x >> 4) + ((int)threadIdx.x >> 4);  // channel 0..B*E-1
  int n = threadIdx.x & 15;
  int b = g / EE;
  int e = g % EE;
  float An = -__expf(A_log[e * NNs + n]);
  float h = 0.f;
  size_t mbase = (size_t)b * LL;
  for (int l = 0; l < LL; ++l) {
    size_t m = mbase + l;
    float dtv = dt[m * EE + e];
    float xv  = bf2f(xc[m * EE + e]);
    float Bn  = dbl[m * 128 + RR + n];
    float Cn  = dbl[m * 128 + RR + NNs + n];
    h = h * __expf(dtv * An) + dtv * xv * Bn;
    float contrib = h * Cn;
    contrib += __shfl_xor(contrib, 8, 16);
    contrib += __shfl_xor(contrib, 4, 16);
    contrib += __shfl_xor(contrib, 2, 16);
    contrib += __shfl_xor(contrib, 1, 16);
    if (n == 0) y[m * EE + e] = contrib;
  }
}

// yg = (y + silu_conv_x * D_skip) * silu(z)   (bf16 out)
__global__ void gate_kernel(const float* __restrict__ y, const u16* __restrict__ xc,
                            const float* __restrict__ xz, const float* __restrict__ Dsk,
                            u16* __restrict__ yg) {
  int idx = blockIdx.x * blockDim.x + threadIdx.x;   // m*EE + e
  int e = idx % EE;
  int m = idx / EE;
  float z  = xz[(size_t)m * (2 * EE) + EE + e];
  float yv = y[idx] + bf2f(xc[idx]) * Dsk[e];
  yg[idx] = f2bf(yv * silu_f(z));
}

// ---------------------------------------------------------------------------
extern "C" void kernel_launch(void* const* d_in, const int* in_sizes, int n_in,
                              void* d_out, int out_size, void* d_ws, size_t ws_size,
                              hipStream_t stream) {
  (void)in_sizes; (void)n_in; (void)out_size; (void)ws_size;

  const int*   tokens   = (const int*)  d_in[0];
  const float* embed    = (const float*)d_in[1];
  const float* norm_w   = (const float*)d_in[2];
  const float* W_in     = (const float*)d_in[3];
  const float* conv_w   = (const float*)d_in[4];
  const float* conv_b   = (const float*)d_in[5];
  const float* W_xproj  = (const float*)d_in[6];
  const float* W_dt     = (const float*)d_in[7];
  const float* dt_bias  = (const float*)d_in[8];
  const float* A_log    = (const float*)d_in[9];
  const float* D_skip   = (const float*)d_in[10];
  const float* W_out    = (const float*)d_in[11];
  const float* fnorm_w  = (const float*)d_in[12];
  const float* lm_head  = (const float*)d_in[13];
  float* out = (float*)d_out;

  // ---- workspace layout (ws ~110 MB) ----
  char* ws = (char*)d_ws;
  float* buf_x    = (float*)(ws);                                  // M*D f32      16.8 MB
  u16*   buf_xn   = (u16*)  (ws + 16777216);                       // M*D bf16      8.4 MB
  u16*   buf_w    = (u16*)  (ws + 25165824);                       // weights bf16 65.5 MB (max: lm_head)
  u16*   buf_xc   = (u16*)  (ws + 90701824);                       // M*E bf16     16.8 MB
  float* buf_dbl  = (float*)(ws + 107479040);                      // M*128 f32     2.1 MB
  u16*   buf_dtin = (u16*)  (ws + 109576192);                      // M*R bf16      0.5 MB

  // ---- big fp32 intermediates live in d_out (524 MB), overwritten by final GEMM ----
  char* os = (char*)d_out;
  float* buf_xz = (float*)(os);                                    // M*4096 f32   64 MB
  float* buf_dt = (float*)(os + 67108864);                         // M*E f32      32 MB
  float* buf_y  = (float*)(os + 100663296);                        // M*E f32      32 MB
  u16*   buf_yg = (u16*)buf_dt;                                    // reuse dt region after scan

  dim3 blk(256);

  // x = embed[tokens]
  embed_kernel<<<(MM * DD) / 256, blk, 0, stream>>>(tokens, embed, buf_x);

  for (int l = 0; l < 2; ++l) {
    const float* nw  = norm_w  + (size_t)l * DD;
    const float* wi  = W_in    + (size_t)l * DD * 2 * EE;
    const float* cw  = conv_w  + (size_t)l * EE * KK;
    const float* cb  = conv_b  + (size_t)l * EE;
    const float* wxp = W_xproj + (size_t)l * EE * 96;
    const float* wdt = W_dt    + (size_t)l * RR * EE;
    const float* dtb = dt_bias + (size_t)l * EE;
    const float* al  = A_log   + (size_t)l * EE * NNs;
    const float* dsk = D_skip  + (size_t)l * EE;
    const float* wo  = W_out   + (size_t)l * EE * DD;

    // xn = bf16(rmsnorm(x))
    rmsnorm_kernel<<<MM, blk, 0, stream>>>(buf_x, nw, buf_xn, DD);

    // xz = xn @ W_in   (4096x1024 . 1024x4096)
    convT_kernel<<<(4096 * 1024) / 256, blk, 0, stream>>>(buf_w, wi, DD, 2 * EE, 4096);
    gemm_bf16_wmma<<<dim3(4096 / 128, MM / 128), blk, 0, stream>>>(
        buf_xn, buf_w, buf_xz, DD, DD, DD, 4096, 0);

    // xc = bf16(silu(causal_conv(xz[:, :E])))
    conv_silu_kernel<<<(MM * EE) / 256, blk, 0, stream>>>(buf_xz, cw, cb, buf_xc);

    // dbl = xc @ W_xproj   (N=96 padded to 128)
    convT_kernel<<<(128 * 2048) / 256, blk, 0, stream>>>(buf_w, wxp, EE, 96, 128);
    gemm_bf16_wmma<<<dim3(128 / 128, MM / 128), blk, 0, stream>>>(
        buf_xc, buf_w, buf_dbl, EE, EE, EE, 128, 0);

    // dt = softplus(dbl[:, :R] @ W_dt + dt_bias)
    slice_kernel<<<(MM * RR) / 256, blk, 0, stream>>>(buf_dtin, buf_dbl);
    convT_kernel<<<(2048 * 64) / 256, blk, 0, stream>>>(buf_w, wdt, RR, EE, EE);
    gemm_bf16_wmma<<<dim3(EE / 128, MM / 128), blk, 0, stream>>>(
        buf_dtin, buf_w, buf_dt, RR, RR, RR, EE, 0);
    softplus_kernel<<<(MM * EE) / 256, blk, 0, stream>>>(buf_dt, dtb);

    // selective scan -> y
    scan_kernel<<<(BB * EE) / 16, blk, 0, stream>>>(buf_dt, buf_xc, buf_dbl, al, buf_y);

    // gate: yg = bf16((y + xc*D_skip) * silu(z))
    gate_kernel<<<(MM * EE) / 256, blk, 0, stream>>>(buf_y, buf_xc, buf_xz, dsk, buf_yg);

    // x += yg @ W_out
    convT_kernel<<<(1024 * 2048) / 256, blk, 0, stream>>>(buf_w, wo, EE, DD, DD);
    gemm_bf16_wmma<<<dim3(DD / 128, MM / 128), blk, 0, stream>>>(
        buf_yg, buf_w, buf_x, EE, EE, EE, DD, 1);
  }

  // final norm + lm_head: out = rmsnorm(x) @ lm_head^T  (lm_head is already N x K)
  rmsnorm_kernel<<<MM, blk, 0, stream>>>(buf_x, fnorm_w, buf_xn, DD);
  conv_direct_kernel<<<((size_t)VV * DD) / 256, blk, 0, stream>>>(buf_w, lm_head, (size_t)VV * DD);
  gemm_bf16_wmma<<<dim3(VV / 128, MM / 128), blk, 0, stream>>>(
      buf_xn, buf_w, out, DD, DD, DD, VV, 0);
}
